// CMAModel_32263794327812
// MI455X (gfx1250) — matrix operand order repaired and checked by reading, compiled, not run on hardware
//
#include <hip/hip_runtime.h>
#include <hip/hip_bf16.h>
#include <stdint.h>

#define DINLINE __device__ __forceinline__

typedef __attribute__((ext_vector_type(16))) __bf16        v16bf;
typedef __attribute__((ext_vector_type(8)))  float         v8f;
typedef __attribute__((ext_vector_type(4)))  unsigned int  u32x4;

union Frag { u32x4 q[2]; v16bf v; };   // 32 bytes: one WMMA bf16 A/B operand

// ---------------- problem dims ----------------
constexpr int BATCH = 2;
constexpr int T     = 2048;
constexpr int C     = 1024;
constexpr int NH    = 16;
constexpr int HD    = 64;
constexpr int MM    = 256;
constexpr int SR    = T + 2 * MM;      // 2560 kv rows per batch
constexpr int ROWS  = BATCH * SR;      // 5120
constexpr int KCAN  = 4;

// ---------------- scalar helpers ----------------
DINLINE unsigned short f2bf(float f) {
  union { float f; unsigned int u; } v; v.f = f;
  unsigned int r = v.u + 0x7fffu + ((v.u >> 16) & 1u);   // round-to-nearest-even
  return (unsigned short)(r >> 16);
}
DINLINE float bf2f(unsigned short h) {
  union { unsigned int u; float f; } v; v.u = ((unsigned int)h) << 16;
  return v.f;
}
DINLINE v8f wmma_bf16(const v16bf& a, const v16bf& b, const v8f& c) {
  // D = A(16x32) * B(32x16) + C, f32 accumulate
  return __builtin_amdgcn_wmma_f32_16x16x32_bf16(false, a, false, b, (short)0, c, false, false);
}
DINLINE float rmax16(float v) {
  #pragma unroll
  for (int m = 1; m < 16; m <<= 1) v = fmaxf(v, __shfl_xor(v, m, 32));
  return v;
}
DINLINE float rsum16(float v) {
  #pragma unroll
  for (int m = 1; m < 16; m <<= 1) v += __shfl_xor(v, m, 32);
  return v;
}

// ---------------- elementwise kernels ----------------
__global__ void cvt_bf16(const float* __restrict__ in, unsigned short* __restrict__ out, int n) {
  int i = blockIdx.x * blockDim.x + threadIdx.x;
  if (i < n) out[i] = f2bf(in[i]);
}

// rows [0,T): x ; [T,T+M): fwd_mem ; [T+M,SR): rev_mem  (per batch), bf16
__global__ void build_xkv(const float* __restrict__ x, const float* __restrict__ fwd,
                          const float* __restrict__ rev, unsigned short* __restrict__ out) {
  int i = blockIdx.x * blockDim.x + threadIdx.x;
  if (i >= ROWS * C) return;
  int c = i % C;
  int r = (i / C) % SR;
  int b = i / (C * SR);
  float v;
  if (r < T)           v = x[((size_t)b * T + r) * C + c];
  else if (r < T + MM) v = fwd[((size_t)b * MM + (r - T)) * C + c];
  else                 v = rev[((size_t)b * MM + (r - T - MM)) * C + c];
  out[i] = f2bf(v);
}

// y + depthwise causal conv(K=4) + bias  ->  bf16
__global__ void canon_kernel(const float* __restrict__ y, const float* __restrict__ cw,
                             const float* __restrict__ cb, unsigned short* __restrict__ out) {
  int i = blockIdx.x * blockDim.x + threadIdx.x;
  if (i >= BATCH * T * C) return;
  int c = i % C;
  int t = (i / C) % T;
  float conv = 0.f;
  #pragma unroll
  for (int j = 0; j < KCAN; ++j) {
    int tt = t - (KCAN - 1) + j;
    if (tt >= 0) conv += y[i + (tt - t) * C] * cw[c * KCAN + j];
  }
  out[i] = f2bf(y[i] + conv + cb[c]);
}

// g[b,t,h] = sigmoid(q[b,t,:] . gate_w[h,:] + gate_b[h]); 16 lanes per head
__global__ __launch_bounds__(256)
void gate_kernel(const unsigned short* __restrict__ q, const float* __restrict__ gw,
                 const float* __restrict__ gb, float* __restrict__ g) {
  const int bt = blockIdx.x;                 // grid = BATCH*T
  const int b = bt / T, t = bt % T;
  const int h = threadIdx.x >> 4;
  const int j = threadIdx.x & 15;
  const unsigned short* qrow = q + (size_t)(b * SR + t) * C;
  const float* wrow = gw + (size_t)h * C;
  float s = 0.f;
  for (int c = j; c < C; c += 16) s += bf2f(qrow[c]) * wrow[c];
  #pragma unroll
  for (int m = 1; m < 16; m <<= 1) s += __shfl_xor(s, m, 32);
  if (j == 0) g[(size_t)bt * NH + h] = 1.0f / (1.0f + __expf(-(s + gb[h])));
}

// ---------------- generic WMMA GEMM: out[m,n] = sum_k A[m,k]*W[n,k] ----------------
// A: Mrows x C bf16 row-major, W: C x C bf16 row-major. Wave tile = 32x64.
__global__ __launch_bounds__(256)
void gemm_bf16_nt(const unsigned short* __restrict__ A, const unsigned short* __restrict__ W,
                  float* __restrict__ outF, unsigned short* __restrict__ outH, int Mrows) {
  const int lane = threadIdx.x & 31;
  const int wave = threadIdx.x >> 5;
  const int Ntiles = C / 64;
  const int total  = (Mrows / 32) * Ntiles;
  int tile = blockIdx.x * 8 + wave;
  if (tile >= total) return;
  const int m0 = (tile / Ntiles) * 32;
  const int n0 = (tile % Ntiles) * 64;
  const int col  = lane & 15;
  const int kloA = (lane < 16) ? 0 : 16;   // A frag: K 0..15 vs 16..31
  const int kloB = (lane < 16) ? 0 : 16;   // B frag: K 0..15 vs 16..31

  v8f z = {0, 0, 0, 0, 0, 0, 0, 0};
  v8f acc[2][4];
  #pragma unroll
  for (int mi = 0; mi < 2; ++mi)
    #pragma unroll
    for (int ni = 0; ni < 4; ++ni) acc[mi][ni] = z;

  const unsigned short* arow0 = A + (size_t)(m0 + col) * C + kloA;
  const unsigned short* arow1 = A + (size_t)(m0 + 16 + col) * C + kloA;
  const unsigned short* wrowb = W + (size_t)(n0 + col) * C + kloB;

  for (int k0 = 0; k0 < C; k0 += 32) {
    __builtin_prefetch(arow0 + k0 + 128, 0, 1);
    Frag a[2], bw[4];
    a[0].q[0] = *(const u32x4*)(arow0 + k0);
    a[0].q[1] = *(const u32x4*)(arow0 + k0 + 8);
    a[1].q[0] = *(const u32x4*)(arow1 + k0);
    a[1].q[1] = *(const u32x4*)(arow1 + k0 + 8);
    #pragma unroll
    for (int ni = 0; ni < 4; ++ni) {
      const unsigned short* wp = wrowb + (size_t)ni * 16 * C + k0;
      bw[ni].q[0] = *(const u32x4*)wp;
      bw[ni].q[1] = *(const u32x4*)(wp + 8);
    }
    #pragma unroll
    for (int mi = 0; mi < 2; ++mi)
      #pragma unroll
      for (int ni = 0; ni < 4; ++ni)
        acc[mi][ni] = wmma_bf16(a[mi].v, bw[ni].v, acc[mi][ni]);
  }

  const int rbase = (lane < 16) ? 0 : 8;   // C-layout: vgpr r -> row rbase+r, lane%16 -> col
  #pragma unroll
  for (int mi = 0; mi < 2; ++mi)
    #pragma unroll
    for (int ni = 0; ni < 4; ++ni)
      #pragma unroll
      for (int r = 0; r < 8; ++r) {
        size_t row = (size_t)m0 + mi * 16 + rbase + r;
        size_t cc  = (size_t)n0 + ni * 16 + col;
        float val = acc[mi][ni][r];
        if (outH) outH[row * C + cc] = f2bf(val);
        else      outF[row * C + cc] = val;
      }
}

// ---------------- flash attention (wave32, WMMA) ----------------
DINLINE void attn_tile(int b, int h, int t0, int s0, bool domask, bool tomem,
                       const Frag aq[2],
                       const unsigned short* __restrict__ k,
                       const unsigned short* __restrict__ v,
                       float m_i[8], float l_i[8], v8f accC[4], v8f accM[4],
                       unsigned short (&p_sh)[16][32], unsigned short (&vT)[HD][32],
                       int lane, int col, int rbase, int kloA, int kloB) {
  v8f z = {0, 0, 0, 0, 0, 0, 0, 0};
  v8f sc[2] = {z, z};
  // scores: 16x32 tile (two 16x16 C tiles), K = HD = 64 -> two k-subtiles each
  #pragma unroll
  for (int ns = 0; ns < 2; ++ns)
    #pragma unroll
    for (int ks = 0; ks < 2; ++ks) {
      Frag bk;  // B[k=d][n=s] = K[s][d] -> contiguous in d
      const unsigned short* kp =
          k + (size_t)(b * SR + s0 + ns * 16 + col) * C + h * HD + ks * 32 + kloB;
      bk.q[0] = *(const u32x4*)kp;
      bk.q[1] = *(const u32x4*)(kp + 8);
      sc[ns] = wmma_bf16(aq[ks].v, bk.v, sc[ns]);
    }

  float pv[2][8];
  #pragma unroll
  for (int ns = 0; ns < 2; ++ns)
    #pragma unroll
    for (int r = 0; r < 8; ++r) {
      float val = sc[ns][r] * 0.125f;                 // 1/sqrt(64)
      if (domask) {
        int t = t0 + rbase + r;
        int s = s0 + ns * 16 + col;
        if (s > t) val = -3.0e38f;
      }
      pv[ns][r] = val;
    }

  float alpha[8];
  #pragma unroll
  for (int r = 0; r < 8; ++r) {
    float nm = fmaxf(m_i[r], rmax16(fmaxf(pv[0][r], pv[1][r])));
    alpha[r] = __expf(m_i[r] - nm);
    m_i[r] = nm;
  }
  #pragma unroll
  for (int r = 0; r < 8; ++r) {
    float p0 = __expf(pv[0][r] - m_i[r]);
    float p1 = __expf(pv[1][r] - m_i[r]);
    p_sh[rbase + r][col]      = f2bf(p0);
    p_sh[rbase + r][16 + col] = f2bf(p1);
    l_i[r] = l_i[r] * alpha[r] + rsum16(p0 + p1);
  }
  v8f av;
  #pragma unroll
  for (int r = 0; r < 8; ++r) av[r] = alpha[r];
  #pragma unroll
  for (int c = 0; c < 4; ++c) { accC[c] *= av; accM[c] *= av; }  // shared normalization

  // stage V^T tile (32 x 64 -> vT[d][s]) so B fragments read contiguously
  {
    union { u32x4 q[4]; unsigned short s[64]; } vu;
    const unsigned short* vrow = v + (size_t)(b * SR + s0 + lane) * C + h * HD;
    #pragma unroll
    for (int i = 0; i < 4; ++i) vu.q[i] = *(const u32x4*)(vrow + i * 16);
    #pragma unroll
    for (int d = 0; d < HD; ++d) vT[d][lane] = vu.s[d];
  }
  __syncthreads();

  Frag ap;  // P as A matrix (16 x 32)
  ap.q[0] = *(const u32x4*)&p_sh[col][kloA];
  ap.q[1] = *(const u32x4*)&p_sh[col][kloA + 8];
  v8f* acc = tomem ? accM : accC;
  #pragma unroll
  for (int c = 0; c < 4; ++c) {
    Frag bv;  // B[k=s][n=d] = vT[d][s]
    const unsigned short* vp = &vT[c * 16 + col][kloB];
    bv.q[0] = *(const u32x4*)vp;
    bv.q[1] = *(const u32x4*)(vp + 8);
    acc[c] = wmma_bf16(ap.v, bv.v, acc[c]);
  }
  __syncthreads();
}

__global__ __launch_bounds__(32)
void attn_kernel(const unsigned short* __restrict__ q, const unsigned short* __restrict__ k,
                 const unsigned short* __restrict__ v, const float* __restrict__ g,
                 float* __restrict__ attn) {
  __shared__ __align__(16) unsigned short p_sh[16][32];
  __shared__ __align__(16) unsigned short vT[HD][32];

  const int lane = threadIdx.x;
  int id = blockIdx.x;
  const int tt = id % (T / 16); id /= (T / 16);
  const int h  = id % NH;       id /= NH;
  const int b  = id;
  const int t0 = tt * 16;
  const int col   = lane & 15;
  const int rbase = (lane < 16) ? 0 : 8;
  const int kloA  = (lane < 16) ? 0 : 16;
  const int kloB  = (lane < 16) ? 0 : 16;

  Frag aq[2];  // q tile 16 x 64 as two A fragments
  {
    const unsigned short* qrow = q + (size_t)(b * SR + t0 + col) * C + h * HD;
    #pragma unroll
    for (int s = 0; s < 2; ++s) {
      const unsigned short* p = qrow + s * 32 + kloA;
      aq[s].q[0] = *(const u32x4*)p;
      aq[s].q[1] = *(const u32x4*)(p + 8);
    }
  }

  v8f z = {0, 0, 0, 0, 0, 0, 0, 0};
  v8f accC[4], accM[4];
  #pragma unroll
  for (int c = 0; c < 4; ++c) { accC[c] = z; accM[c] = z; }
  float m_i[8], l_i[8];
  #pragma unroll
  for (int r = 0; r < 8; ++r) { m_i[r] = -3.0e38f; l_i[r] = 0.0f; }

  for (int s0 = 0; s0 < t0 + 16; s0 += 32)        // causal region
    attn_tile(b, h, t0, s0, (s0 + 31 > t0), false, aq, k, v,
              m_i, l_i, accC, accM, p_sh, vT, lane, col, rbase, kloA, kloB);
  for (int s0 = T; s0 < SR; s0 += 32)             // memory region (unmasked, gated)
    attn_tile(b, h, t0, s0, false, true, aq, k, v,
              m_i, l_i, accC, accM, p_sh, vT, lane, col, rbase, kloA, kloB);

  float gr[8], inv[8];
  #pragma unroll
  for (int r = 0; r < 8; ++r) {
    int t = t0 + rbase + r;
    gr[r]  = g[((size_t)(b * T + t)) * NH + h];
    inv[r] = 1.0f / l_i[r];
  }
  #pragma unroll
  for (int c = 0; c < 4; ++c)
    #pragma unroll
    for (int r = 0; r < 8; ++r) {
      int t = t0 + rbase + r;
      float val = (accC[c][r] + gr[r] * accM[c][r]) * inv[r];
      attn[((size_t)(b * T + t)) * C + h * HD + c * 16 + col] = val;
    }
}

// ---------------- host launcher ----------------
extern "C" void kernel_launch(void* const* d_in, const int* in_sizes, int n_in,
                              void* d_out, int out_size, void* d_ws, size_t ws_size,
                              hipStream_t stream) {
  (void)in_sizes; (void)n_in; (void)out_size; (void)ws_size;
  const float* x   = (const float*)d_in[0];
  const float* fwd = (const float*)d_in[1];
  const float* rev = (const float*)d_in[2];
  const float* Wq  = (const float*)d_in[3];
  const float* Wk  = (const float*)d_in[4];
  const float* Wv  = (const float*)d_in[5];
  const float* Wo  = (const float*)d_in[6];
  const float* gw  = (const float*)d_in[7];
  const float* gb  = (const float*)d_in[8];
  const float* cw  = (const float*)d_in[9];
  const float* cb  = (const float*)d_in[10];

  char* ws = (char*)d_ws;
  size_t off = 0;
  auto take = [&](size_t bytes) -> char* {
    char* p = ws + off;
    off = (off + bytes + 255) & ~(size_t)255;
    return p;
  };
  unsigned short* xkv = (unsigned short*)take((size_t)ROWS * C * 2);
  unsigned short* wqh = (unsigned short*)take((size_t)C * C * 2);
  unsigned short* wkh = (unsigned short*)take((size_t)C * C * 2);
  unsigned short* wvh = (unsigned short*)take((size_t)C * C * 2);
  unsigned short* woh = (unsigned short*)take((size_t)C * C * 2);
  unsigned short* qh  = (unsigned short*)take((size_t)ROWS * C * 2);
  unsigned short* kh  = (unsigned short*)take((size_t)ROWS * C * 2);
  unsigned short* vh  = (unsigned short*)take((size_t)ROWS * C * 2);
  float*          gbf = (float*)take((size_t)BATCH * T * NH * 4);
  float*          atf = (float*)take((size_t)BATCH * T * C * 4);
  unsigned short* a2h = (unsigned short*)take((size_t)BATCH * T * C * 2);

  const int thr = 256;
  const int nW = C * C;
  cvt_bf16<<<(nW + thr - 1) / thr, thr, 0, stream>>>(Wq, wqh, nW);
  cvt_bf16<<<(nW + thr - 1) / thr, thr, 0, stream>>>(Wk, wkh, nW);
  cvt_bf16<<<(nW + thr - 1) / thr, thr, 0, stream>>>(Wv, wvh, nW);
  cvt_bf16<<<(nW + thr - 1) / thr, thr, 0, stream>>>(Wo, woh, nW);
  const int nX = ROWS * C;
  build_xkv<<<(nX + thr - 1) / thr, thr, 0, stream>>>(x, fwd, rev, xkv);

  auto gemm_blocks = [](int Mrows) { int tiles = (Mrows / 32) * (C / 64); return (tiles + 7) / 8; };
  gemm_bf16_nt<<<gemm_blocks(ROWS), 256, 0, stream>>>(xkv, wqh, nullptr, qh, ROWS);
  gemm_bf16_nt<<<gemm_blocks(ROWS), 256, 0, stream>>>(xkv, wkh, nullptr, kh, ROWS);
  gemm_bf16_nt<<<gemm_blocks(ROWS), 256, 0, stream>>>(xkv, wvh, nullptr, vh, ROWS);

  gate_kernel<<<BATCH * T, 256, 0, stream>>>(qh, gw, gb, gbf);
  attn_kernel<<<BATCH * NH * (T / 16), 32, 0, stream>>>(qh, kh, vh, gbf, atf);

  const int nY = BATCH * T * C;
  canon_kernel<<<(nY + thr - 1) / thr, thr, 0, stream>>>(atf, cw, cb, a2h);

  gemm_bf16_nt<<<gemm_blocks(BATCH * T), 256, 0, stream>>>(a2h, woh, (float*)d_out, nullptr, BATCH * T);
}
